// Cascade_CNN_RNN_29695403885014
// MI455X (gfx1250) — matrix operand order repaired and checked by reading, not compiled, not run on hardware
//
#include <hip/hip_runtime.h>

typedef _Float16 h16;
typedef _Float16 v16h __attribute__((ext_vector_type(16)));
typedef float    v8f  __attribute__((ext_vector_type(8)));

union Frag16 { v16h v; unsigned u[8]; h16 h[16]; };

__device__ __forceinline__ float clamp01(float x) { return x < 0.f ? 0.f : (x > 1.f ? 1.f : x); }
__device__ __forceinline__ float clamp11(float x) { return x < -1.f ? -1.f : (x > 1.f ? 1.f : x); }

#define HW   110   // 10*11 pixels
#define C1   16
#define C2   32
#define K2   160   // conv2 GEMM K: 16*9=144 padded to 160
#define P2   112   // pixels padded to 112
#define F3IN 3520  // 32*110, divisible by 32
#define FC   256
#define HID  64
#define WIN  10
#define NC   7

// ---------------------------------------------------------------- kernel 0
__global__ __launch_bounds__(256) void prep_weights(
    const float* __restrict__ w2, const float* __restrict__ w3,
    const float* __restrict__ wf, const float* __restrict__ wn,
    h16* __restrict__ w2p, h16* __restrict__ w3h,
    h16* __restrict__ wfh, h16* __restrict__ wnh)
{
  const int N0 = C2 * K2;          // 5120
  const int N1 = FC * F3IN;        // 901120
  const int N2 = HID * (FC + HID); // 20480
  const int total = N0 + N1 + 2 * N2;
  for (int i = blockIdx.x * blockDim.x + threadIdx.x; i < total;
       i += gridDim.x * blockDim.x) {
    if (i < N0) {
      int m = i / K2, k = i % K2;
      w2p[i] = (k < 144) ? (h16)w2[m * 144 + k] : (h16)0.f;
    } else if (i < N0 + N1) {
      int j = i - N0; w3h[j] = (h16)w3[j];
    } else if (i < N0 + N1 + N2) {
      int j = i - N0 - N1; wfh[j] = (h16)wf[j];
    } else {
      int j = i - N0 - N1 - N2; wnh[j] = (h16)wn[j];
    }
  }
}

// ---------------------------------------------------------------- kernel 1
// One workgroup per frame: jitter + conv1 (VALU) -> im2col in LDS ->
// conv2 as WMMA GEMM  D[32x112] = W2[32x160] x im2colT[160x112].
__global__ __launch_bounds__(128) void conv12_fused(
    const float* __restrict__ inp, const float* __restrict__ noise,
    const float* __restrict__ w1, const h16* __restrict__ w2p,
    h16* __restrict__ x2h)
{
  __shared__ float s_in[HW];
  __shared__ float s_w1[C1 * 9];
  __shared__ __align__(16) h16 s_a1[C1 * HW];
  __shared__ __align__(16) h16 s_col[P2 * K2];   // [pixel][k], 35840 B

  const int f = blockIdx.x;
  const int tid = threadIdx.x;

  if (tid < HW) {
    float v = inp[(size_t)f * HW + tid];
    float nz = noise[(size_t)f * HW + tid];
    s_in[tid] = v * (1.f + nz * 0.05f);     // x * (1 + U/20)
  }
  for (int i = tid; i < C1 * 9; i += 128) s_w1[i] = w1[i];
  __syncthreads();

  // conv1 + ScaleReLU -> f16 in LDS
  for (int idx = tid; idx < C1 * HW; idx += 128) {
    int c = idx / HW, p = idx % HW;
    int y = p / 11, x = p % 11;
    float acc = 0.f;
    #pragma unroll
    for (int dy = 0; dy < 3; ++dy) {
      int yy = y + dy - 1;
      if (yy < 0 || yy >= 10) continue;
      #pragma unroll
      for (int dx = 0; dx < 3; ++dx) {
        int xx = x + dx - 1;
        if (xx < 0 || xx >= 11) continue;
        acc += s_in[yy * 11 + xx] * s_w1[c * 9 + dy * 3 + dx];
      }
    }
    s_a1[idx] = (h16)clamp01(acc);
  }
  __syncthreads();

  // transposed im2col: s_col[p][k], k = c*9 + dy*3 + dx (zero padded)
  for (int idx = tid; idx < P2 * K2; idx += 128) {
    int p = idx / K2, k = idx % K2;
    h16 val = (h16)0.f;
    if (p < HW && k < 144) {
      int c = k / 9, r9 = k % 9;
      int dy = r9 / 3, dx = r9 % 3;
      int y = p / 11, x = p % 11;
      int yy = y + dy - 1, xx = x + dx - 1;
      if (yy >= 0 && yy < 10 && xx >= 0 && xx < 11)
        val = s_a1[c * HW + yy * 11 + xx];
    }
    s_col[idx] = val;
  }
  __syncthreads();

  const int wave = tid >> 5, lane = tid & 31;
  const int lhalf = (lane >= 16) ? 1 : 0;
  const int l15 = lane & 15;

  for (int tile = wave; tile < 14; tile += 4) {  // 2 m-tiles x 7 n-tiles
    int mt = tile / 7, nt = tile % 7;
    v8f acc = {};
    #pragma unroll
    for (int kt = 0; kt < 5; ++kt) {
      Frag16 a, b;
      int m = mt * 16 + l15;
      #pragma unroll
      for (int r = 0; r < 8; ++r) {
        int ka = kt * 32 + ((r < 4) ? 0 : 16) + lhalf * 8 + (r & 3) * 2;
        a.u[r] = *reinterpret_cast<const unsigned*>(w2p + m * K2 + ka);
      }
      int n = nt * 16 + l15;
      #pragma unroll
      for (int r = 0; r < 8; ++r) {
        int kb = kt * 32 + ((r < 4) ? 0 : 16) + lhalf * 8 + (r & 3) * 2;
        b.u[r] = *reinterpret_cast<const unsigned*>(&s_col[n * K2 + kb]);
      }
      acc = __builtin_amdgcn_wmma_f32_16x16x32_f16(false, a.v, false, b.v,
                                                   (short)0, acc, false, false);
    }
    #pragma unroll
    for (int r = 0; r < 8; ++r) {
      int m = mt * 16 + r + lhalf * 8;   // output channel
      int n = nt * 16 + l15;             // pixel
      if (n < HW)
        x2h[(size_t)f * F3IN + m * HW + n] = (h16)clamp01(acc[r]);
    }
  }
}

// ---------------------------------------------------------------- kernel 2
// fc3: out[NF x 256] = X[NF x 3520] * W3^T, f16 WMMA, f32 accumulate.
// One wave per (16-frame m-tile, 8 n-tiles), A fragment reused 8x.
__global__ __launch_bounds__(256) void fc3_gemm(
    const h16* __restrict__ x2h, const h16* __restrict__ w3h,
    h16* __restrict__ x3h)
{
  const int gwave = (blockIdx.x * 256 + threadIdx.x) >> 5;
  const int lane = threadIdx.x & 31;
  const int lhalf = (lane >= 16) ? 1 : 0;
  const int l15 = lane & 15;
  const int mtile = gwave >> 1;
  const int nhalf = gwave & 1;
  const int f0 = mtile * 16;

  v8f acc[8] = {};
  const size_t arow = (size_t)(f0 + l15) * F3IN;

  for (int kt = 0; kt < F3IN / 32; ++kt) {   // 110 K-tiles
    Frag16 a;
    #pragma unroll
    for (int r = 0; r < 8; ++r) {
      int ka = kt * 32 + ((r < 4) ? 0 : 16) + lhalf * 8 + (r & 3) * 2;
      a.u[r] = *reinterpret_cast<const unsigned*>(x2h + arow + ka);
    }
    #pragma unroll
    for (int j = 0; j < 8; ++j) {
      int n = (nhalf * 8 + j) * 16 + l15;
      Frag16 b;
      #pragma unroll
      for (int r = 0; r < 8; ++r) {
        int kb = kt * 32 + ((r < 4) ? 0 : 16) + lhalf * 8 + (r & 3) * 2;
        b.u[r] = *reinterpret_cast<const unsigned*>(w3h + (size_t)n * F3IN + kb);
      }
      acc[j] = __builtin_amdgcn_wmma_f32_16x16x32_f16(false, a.v, false, b.v,
                                                      (short)0, acc[j], false, false);
    }
  }
  #pragma unroll
  for (int j = 0; j < 8; ++j) {
    #pragma unroll
    for (int r = 0; r < 8; ++r) {
      int fr = f0 + r + lhalf * 8;
      int n = (nhalf * 8 + j) * 16 + l15;
      x3h[(size_t)fr * FC + n] = (h16)clamp01(acc[j][r]);
    }
  }
}

// ---------------------------------------------------------------- kernel 3
// MGU over 10 steps; each wave owns 16 sequences, h kept in C/D fragments.
// Gate GEMMs (M=16, N=64, K=320) via WMMA; h / f*h staged through LDS
// for the concat part of the A fragments. Ends with fc5 (7-wide) in VALU.
__global__ __launch_bounds__(256) void mgu_fc5(
    const h16* __restrict__ x3h, const h16* __restrict__ wfh,
    const h16* __restrict__ wnh, const float* __restrict__ w5,
    float* __restrict__ out)
{
  __shared__ __align__(16) h16   s_g[8][16 * HID];   // per-wave: h or f*h (f16)
  __shared__ __align__(16) float s_hf[8][16 * HID];  // per-wave: final h (f32)

  const int wave = threadIdx.x >> 5;
  const int lane = threadIdx.x & 31;
  const int lhalf = (lane >= 16) ? 1 : 0;
  const int l15 = lane & 15;
  const int seq0 = (blockIdx.x * 8 + wave) * 16;

  v8f h[4] = {};

  for (int t = 0; t < WIN; ++t) {
    asm volatile("s_wait_dscnt 0x0" ::: "memory");   // prior-step LDS reads done
    // stage h (f16) for A-fragment consumption (k = 256..319)
    #pragma unroll
    for (int j = 0; j < 4; ++j)
      #pragma unroll
      for (int r = 0; r < 8; ++r) {
        int m = r + lhalf * 8, n = j * 16 + l15;
        s_g[wave][m * HID + n] = (h16)h[j][r];
      }
    asm volatile("s_wait_dscnt 0x0" ::: "memory");

    const size_t xrow = ((size_t)(seq0 + l15) * WIN + t) * FC;

    // ---- forget gate: f = clip([x,h]*wf^T/6 + 0.5, 0, 1)
    v8f af[4] = {};
    #pragma unroll
    for (int kt = 0; kt < 10; ++kt) {
      Frag16 a;
      #pragma unroll
      for (int r = 0; r < 8; ++r) {
        int ka = kt * 32 + ((r < 4) ? 0 : 16) + lhalf * 8 + (r & 3) * 2;
        if (kt < 8)
          a.u[r] = *reinterpret_cast<const unsigned*>(x3h + xrow + ka);
        else
          a.u[r] = *reinterpret_cast<const unsigned*>(&s_g[wave][l15 * HID + (ka - 256)]);
      }
      #pragma unroll
      for (int j = 0; j < 4; ++j) {
        int n = j * 16 + l15;
        Frag16 b;
        #pragma unroll
        for (int r = 0; r < 8; ++r) {
          int kb = kt * 32 + ((r < 4) ? 0 : 16) + lhalf * 8 + (r & 3) * 2;
          b.u[r] = *reinterpret_cast<const unsigned*>(wfh + n * (FC + HID) + kb);
        }
        af[j] = __builtin_amdgcn_wmma_f32_16x16x32_f16(false, a.v, false, b.v,
                                                       (short)0, af[j], false, false);
      }
    }
    v8f fg[4];
    #pragma unroll
    for (int j = 0; j < 4; ++j)
      #pragma unroll
      for (int r = 0; r < 8; ++r)
        fg[j][r] = clamp01(af[j][r] * (1.f / 6.f) + 0.5f);

    // stage f*h
    asm volatile("s_wait_dscnt 0x0" ::: "memory");
    #pragma unroll
    for (int j = 0; j < 4; ++j)
      #pragma unroll
      for (int r = 0; r < 8; ++r) {
        int m = r + lhalf * 8, n = j * 16 + l15;
        s_g[wave][m * HID + n] = (h16)(fg[j][r] * h[j][r]);
      }
    asm volatile("s_wait_dscnt 0x0" ::: "memory");

    // ---- new gate: n = clip([x, f*h]*wn^T, -1, 1)
    v8f an[4] = {};
    #pragma unroll
    for (int kt = 0; kt < 10; ++kt) {
      Frag16 a;
      #pragma unroll
      for (int r = 0; r < 8; ++r) {
        int ka = kt * 32 + ((r < 4) ? 0 : 16) + lhalf * 8 + (r & 3) * 2;
        if (kt < 8)
          a.u[r] = *reinterpret_cast<const unsigned*>(x3h + xrow + ka);
        else
          a.u[r] = *reinterpret_cast<const unsigned*>(&s_g[wave][l15 * HID + (ka - 256)]);
      }
      #pragma unroll
      for (int j = 0; j < 4; ++j) {
        int n = j * 16 + l15;
        Frag16 b;
        #pragma unroll
        for (int r = 0; r < 8; ++r) {
          int kb = kt * 32 + ((r < 4) ? 0 : 16) + lhalf * 8 + (r & 3) * 2;
          b.u[r] = *reinterpret_cast<const unsigned*>(wnh + n * (FC + HID) + kb);
        }
        an[j] = __builtin_amdgcn_wmma_f32_16x16x32_f16(false, a.v, false, b.v,
                                                       (short)0, an[j], false, false);
      }
    }
    // h = (1-f)*h + f*clip(n,-1,1)
    #pragma unroll
    for (int j = 0; j < 4; ++j)
      #pragma unroll
      for (int r = 0; r < 8; ++r) {
        float nv = clamp11(an[j][r]);
        h[j][r] = (1.f - fg[j][r]) * h[j][r] + fg[j][r] * nv;
      }
  }

  // fc5 + hardtanh
  asm volatile("s_wait_dscnt 0x0" ::: "memory");
  #pragma unroll
  for (int j = 0; j < 4; ++j)
    #pragma unroll
    for (int r = 0; r < 8; ++r) {
      int m = r + lhalf * 8, n = j * 16 + l15;
      s_hf[wave][m * HID + n] = h[j][r];
    }
  asm volatile("s_wait_dscnt 0x0" ::: "memory");

  for (int o = lane; o < 16 * NC; o += 32) {
    int sl = o / NC, c = o % NC;
    float acc = 0.f;
    #pragma unroll 8
    for (int k = 0; k < HID; ++k)
      acc += s_hf[wave][sl * HID + k] * w5[c * HID + k];
    out[(size_t)(seq0 + sl) * NC + c] = clamp11(acc);
  }
}

// ---------------------------------------------------------------- launcher
extern "C" void kernel_launch(void* const* d_in, const int* in_sizes, int n_in,
                              void* d_out, int out_size, void* d_ws, size_t ws_size,
                              hipStream_t stream) {
  const float* inp   = (const float*)d_in[0];
  const float* noise = (const float*)d_in[1];
  const float* w1    = (const float*)d_in[2];
  const float* w2    = (const float*)d_in[3];
  const float* w3    = (const float*)d_in[4];
  const float* wf    = (const float*)d_in[5];
  const float* wn    = (const float*)d_in[6];
  const float* w5    = (const float*)d_in[7];
  float* out = (float*)d_out;

  const int NF = in_sizes[0] / HW;  // 20480 frames
  const int NS = NF / WIN;          // 2048 sequences

  char* ws = (char*)d_ws;
  size_t off = 0;
  auto take = [&](size_t bytes) {
    char* p = ws + off;
    off = (off + bytes + 255) & ~(size_t)255;
    return p;
  };
  h16* w2p = (h16*)take((size_t)C2 * K2 * 2);
  h16* w3h = (h16*)take((size_t)FC * F3IN * 2);
  h16* wfh = (h16*)take((size_t)HID * (FC + HID) * 2);
  h16* wnh = (h16*)take((size_t)HID * (FC + HID) * 2);
  h16* x2h = (h16*)take((size_t)NF * F3IN * 2);  // conv2 acts, fc3-flatten order
  h16* x3h = (h16*)take((size_t)NF * FC * 2);    // fc3 outputs
  (void)ws_size; (void)n_in; (void)out_size;     // ~157 MB total scratch

  const int prepTotal = C2 * K2 + FC * F3IN + 2 * HID * (FC + HID);
  prep_weights<<<(prepTotal + 255) / 256, 256, 0, stream>>>(
      w2, w3, wf, wn, w2p, w3h, wfh, wnh);
  conv12_fused<<<NF, 128, 0, stream>>>(inp, noise, w1, w2p, x2h);
  fc3_gemm<<<((NF / 16) * 2) / 8, 256, 0, stream>>>(x2h, w3h, x3h);
  mgu_fc5<<<NS / 128, 256, 0, stream>>>(x3h, wfh, wnh, w5, out);
}
